// MetaPathGNN_20160576487476
// MI455X (gfx1250) — matrix operand order repaired and checked by reading, not compiled, not run on hardware
//
#include <hip/hip_runtime.h>
#include <hip/hip_bf16.h>

#define N_NODES 100000
#define N_EDGES 600000

typedef __attribute__((ext_vector_type(16))) __bf16 v16bf;
typedef __attribute__((ext_vector_type(8)))  __bf16 v8bf;
typedef __attribute__((ext_vector_type(4)))  __bf16 v4bf;
typedef __attribute__((ext_vector_type(8)))  float  v8f;

// ---------------------------------------------------------------------------
// A fragment: 16x32 (MxK) bf16, row-major tile in LDS, leading dim ld.
// Lane layout: lanes 0-15 M=lane, K base 0;
// lanes 16-31 M=lane-16, base 16.  Per lane the 16 values are contiguous:
// K = kb..kb+7 (VGPR0-3) and kb+8..kb+15 (VGPR4-7).
// -> two ds_load_b128 per fragment.
// ---------------------------------------------------------------------------
__device__ __forceinline__ v16bf lds_load_a(const __bf16* t, int ld, int k0, int lane) {
  const int m  = lane & 15;
  const int kb = k0 + ((lane & 16) ? 16 : 0);
  const v8bf lo = *(const v8bf*)(t + m * ld + kb);        // 16B aligned
  const v8bf hi = *(const v8bf*)(t + m * ld + kb + 8);    // 16B aligned
  return __builtin_shufflevector(lo, hi, 0, 1, 2, 3, 4, 5, 6, 7,
                                         8, 9, 10, 11, 12, 13, 14, 15);
}

// ---------------------------------------------------------------------------
// B fragment: 32x16 (KxN) bf16 from *transposed* weights wT[n][k] in LDS
// (leading dim ldk).  Lanes 0-15: N=lane, K=k0..k0+15 contiguous;
// lanes 16-31: N=lane-16, K=k0+16..k0+31 contiguous.
// -> one contiguous 32-byte load (two ds_load_b128), zero packing VALU.
// ---------------------------------------------------------------------------
__device__ __forceinline__ v16bf lds_load_bt(const __bf16* wt, int ldk, int k0, int n0, int lane) {
  const int n  = n0 + (lane & 15);
  const int kb = k0 + ((lane & 16) ? 16 : 0);
  return *(const v16bf*)(wt + n * ldk + kb);              // 32B aligned
}

// Stage a 16x128 f32 tile from global -> bf16 LDS tile (float4 in, b64 out)
__device__ __forceinline__ void stage_tile(const float* __restrict__ src, __bf16* dst,
                                           int row0, int lane) {
  for (int c = lane; c < 16 * 32; c += 32) {          // 512 float4 chunks
    const int r  = c >> 5;
    const int c4 = (c & 31) * 4;
    int gr = row0 + r; if (gr >= N_NODES) gr = N_NODES - 1;
    const float4 f = *(const float4*)(src + (size_t)gr * 128 + c4);
    v4bf pk; pk[0] = (__bf16)f.x; pk[1] = (__bf16)f.y; pk[2] = (__bf16)f.z; pk[3] = (__bf16)f.w;
    *(v4bf*)(dst + r * 128 + c4) = pk;                // ds_store_b64
  }
}

// Cooperatively stage a [K x 64] f32 weight transposed into LDS as wT[64+][K]
__device__ __forceinline__ void stage_wT(const float* __restrict__ w, __bf16* wt,
                                         int K, int ldk, int kofs, int tid, int nthreads) {
  for (int i = tid * 4; i < K * 64; i += nthreads * 4) {
    const float4 f = *(const float4*)(w + i);         // coalesced
    const int k = i >> 6;                             // i / 64
    const int n = i & 63;
    wt[(n + 0) * ldk + kofs + k] = (__bf16)f.x;
    wt[(n + 1) * ldk + kofs + k] = (__bf16)f.y;
    wt[(n + 2) * ldk + kofs + k] = (__bf16)f.z;
    wt[(n + 3) * ldk + kofs + k] = (__bf16)f.w;
  }
}

// ---------------------------------------------------------------------------
// Kernel 1: fused 3-layer MLP.  128 threads = 4 waves, 16 rows per wave.
//   h = (relu(relu(x@W1+b1)@W2+b2))@W3 + b3     x:[N,128] -> h:[N,128]
// Weights live transposed in LDS: sWnt[n][k].
// ---------------------------------------------------------------------------
__global__ __launch_bounds__(128) void mlp_kernel(
    const float* __restrict__ x,
    const float* __restrict__ w1, const float* __restrict__ b1,
    const float* __restrict__ w2, const float* __restrict__ b2,
    const float* __restrict__ w3, const float* __restrict__ b3,
    float* __restrict__ hout)
{
  __shared__ __bf16 sW1t[64 * 128];    // [n][k] k=0..127
  __shared__ __bf16 sW2t[64 * 64];     // [n][k] k=0..63
  __shared__ __bf16 sW3t[128 * 64];    // [n][k] k=0..63
  __shared__ float  sB1[64], sB2[64], sB3[128];
  __shared__ __bf16 sX[4][16 * 128];   // per-wave input / layer-2 output stage
  __shared__ __bf16 sH[4][16 * 64];    // per-wave layer-1 output stage

  const int tid = threadIdx.x;
  stage_wT(w1, sW1t, 128, 128, 0, tid, 128);
  stage_wT(w2, sW2t, 64, 64, 0, tid, 128);
  // w3 is [64 x 128]: transpose to [128][64]
  for (int i = tid * 4; i < 64 * 128; i += 128 * 4) {
    const float4 f = *(const float4*)(w3 + i);
    const int k = i >> 7;               // i / 128
    const int n = i & 127;
    sW3t[(n + 0) * 64 + k] = (__bf16)f.x;
    sW3t[(n + 1) * 64 + k] = (__bf16)f.y;
    sW3t[(n + 2) * 64 + k] = (__bf16)f.z;
    sW3t[(n + 3) * 64 + k] = (__bf16)f.w;
  }
  if (tid < 64)  { sB1[tid] = b1[tid]; sB2[tid] = b2[tid]; }
  if (tid < 128) sB3[tid] = b3[tid];
  __syncthreads();

  const int wave = tid >> 5;
  const int lane = tid & 31;
  const int row0 = blockIdx.x * 64 + wave * 16;
  const int nlo  = lane & 15;
  const int mhi  = (lane & 16) ? 8 : 0;

  __bf16* xb = sX[wave];
  __bf16* hb = sH[wave];

  stage_tile(x, xb, row0, lane);
  __syncthreads();

  // ---- layer 1: [16x128]@[128x64] + b1, relu -> sH
#pragma unroll
  for (int nt = 0; nt < 4; ++nt) {
    v8f acc;
    const float bv = sB1[nt * 16 + nlo];
#pragma unroll
    for (int v = 0; v < 8; ++v) acc[v] = bv;
#pragma unroll
    for (int ks = 0; ks < 4; ++ks) {
      v16bf a = lds_load_a(xb, 128, ks * 32, lane);
      v16bf b = lds_load_bt(sW1t, 128, ks * 32, nt * 16, lane);
      acc = __builtin_amdgcn_wmma_f32_16x16x32_bf16(false, a, false, b, (short)0, acc, false, false);
    }
#pragma unroll
    for (int v = 0; v < 8; ++v) {
      float f = acc[v] > 0.f ? acc[v] : 0.f;
      hb[(v + mhi) * 64 + nt * 16 + nlo] = (__bf16)f;
    }
  }
  __syncthreads();

  // ---- layer 2: [16x64]@[64x64] + b2, relu -> reuse sX (first 2KB)
#pragma unroll
  for (int nt = 0; nt < 4; ++nt) {
    v8f acc;
    const float bv = sB2[nt * 16 + nlo];
#pragma unroll
    for (int v = 0; v < 8; ++v) acc[v] = bv;
#pragma unroll
    for (int ks = 0; ks < 2; ++ks) {
      v16bf a = lds_load_a(hb, 64, ks * 32, lane);
      v16bf b = lds_load_bt(sW2t, 64, ks * 32, nt * 16, lane);
      acc = __builtin_amdgcn_wmma_f32_16x16x32_bf16(false, a, false, b, (short)0, acc, false, false);
    }
#pragma unroll
    for (int v = 0; v < 8; ++v) {
      float f = acc[v] > 0.f ? acc[v] : 0.f;
      xb[(v + mhi) * 64 + nt * 16 + nlo] = (__bf16)f;
    }
  }
  __syncthreads();

  // ---- layer 3: [16x64]@[64x128] + b3 (no relu) -> global h (f32)
#pragma unroll
  for (int nt = 0; nt < 8; ++nt) {
    v8f acc;
    const float bv = sB3[nt * 16 + nlo];
#pragma unroll
    for (int v = 0; v < 8; ++v) acc[v] = bv;
#pragma unroll
    for (int ks = 0; ks < 2; ++ks) {
      v16bf a = lds_load_a(xb, 64, ks * 32, lane);
      v16bf b = lds_load_bt(sW3t, 64, ks * 32, nt * 16, lane);
      acc = __builtin_amdgcn_wmma_f32_16x16x32_bf16(false, a, false, b, (short)0, acc, false, false);
    }
#pragma unroll
    for (int v = 0; v < 8; ++v) {
      const int gr = row0 + v + mhi;
      if (gr < N_NODES) hout[(size_t)gr * 128 + nt * 16 + nlo] = acc[v];
    }
  }
}

// ---------------------------------------------------------------------------
// Kernel 2: edge scatter  agg[row] += h[col].  One edge per wave; each lane
// moves a float4 (512B/row per wave).  agg (51MB) lives in L2 -> fast atomics.
// ---------------------------------------------------------------------------
__global__ __launch_bounds__(256) void scatter_kernel(
    const float* __restrict__ h, const int* __restrict__ ei, float* __restrict__ agg)
{
  const int gw   = (int)((blockIdx.x * 256u + threadIdx.x) >> 5);
  const int lane = threadIdx.x & 31;
  if (gw >= N_EDGES) return;
  const int r = ei[gw];             // edge_index[0][0][e]
  const int c = ei[N_EDGES + gw];   // edge_index[0][1][e]
  const float4 v = *reinterpret_cast<const float4*>(h + (size_t)c * 128 + lane * 4);
  float* dst = agg + (size_t)r * 128 + lane * 4;
  unsafeAtomicAdd(dst + 0, v.x);
  unsafeAtomicAdd(dst + 1, v.y);
  unsafeAtomicAdd(dst + 2, v.z);
  unsafeAtomicAdd(dst + 3, v.w);
}

// ---------------------------------------------------------------------------
// Kernel 3: out = relu([agg | h | x] @ [wl;w0;w1] + (wlb+w0b+w1b))
// Single K=384 GEMM; weights transposed in LDS as sWt[n][384].
// ---------------------------------------------------------------------------
__global__ __launch_bounds__(128) void out_kernel(
    const float* __restrict__ agg, const float* __restrict__ h, const float* __restrict__ x,
    const float* __restrict__ wl, const float* __restrict__ wlb,
    const float* __restrict__ w0, const float* __restrict__ w0b,
    const float* __restrict__ w1, const float* __restrict__ w1b,
    float* __restrict__ out)
{
  __shared__ __bf16 sWt[64 * 384];     // [n][k], k: 0-127 wl | 128-255 w0 | 256-383 w1
  __shared__ float  sB[64];
  __shared__ __bf16 sA[4][16 * 128];   // per-wave stage

  const int tid = threadIdx.x;
  stage_wT(wl, sWt, 128, 384, 0,   tid, 128);
  stage_wT(w0, sWt, 128, 384, 128, tid, 128);
  stage_wT(w1, sWt, 128, 384, 256, tid, 128);
  if (tid < 64) sB[tid] = wlb[tid] + w0b[tid] + w1b[tid];
  __syncthreads();

  const int wave = tid >> 5;
  const int lane = tid & 31;
  const int row0 = blockIdx.x * 64 + wave * 16;
  const int nlo  = lane & 15;
  const int mhi  = (lane & 16) ? 8 : 0;
  __bf16* ab = sA[wave];

  v8f acc[4];
#pragma unroll
  for (int nt = 0; nt < 4; ++nt) {
    const float bv = sB[nt * 16 + nlo];
#pragma unroll
    for (int v = 0; v < 8; ++v) acc[nt][v] = bv;
  }

  const float* srcs[3] = { agg, h, x };
#pragma unroll
  for (int seg = 0; seg < 3; ++seg) {
    stage_tile(srcs[seg], ab, row0, lane);
    __syncthreads();
#pragma unroll
    for (int nt = 0; nt < 4; ++nt) {
#pragma unroll
      for (int ks = 0; ks < 4; ++ks) {
        v16bf a = lds_load_a(ab, 128, ks * 32, lane);
        v16bf b = lds_load_bt(sWt, 384, seg * 128 + ks * 32, nt * 16, lane);
        acc[nt] = __builtin_amdgcn_wmma_f32_16x16x32_bf16(false, a, false, b, (short)0, acc[nt], false, false);
      }
    }
    __syncthreads();
  }

#pragma unroll
  for (int nt = 0; nt < 4; ++nt) {
#pragma unroll
    for (int v = 0; v < 8; ++v) {
      const int gr = row0 + v + mhi;
      const float f = acc[nt][v] > 0.f ? acc[nt][v] : 0.f;
      if (gr < N_NODES) out[(size_t)gr * 64 + nt * 16 + nlo] = f;
    }
  }
}

// ---------------------------------------------------------------------------
extern "C" void kernel_launch(void* const* d_in, const int* in_sizes, int n_in,
                              void* d_out, int out_size, void* d_ws, size_t ws_size,
                              hipStream_t stream)
{
  const float* x    = (const float*)d_in[0];
  const int*   ei   = (const int*)d_in[1];   // [3,2,E] flat; rel 0 -> rows @0, cols @E
  // d_in[2] edge_type: unused by the reference (metapath selects relation 0)
  const float* mw1  = (const float*)d_in[3];
  const float* mb1  = (const float*)d_in[4];
  const float* mw2  = (const float*)d_in[5];
  const float* mb2  = (const float*)d_in[6];
  const float* mw3  = (const float*)d_in[7];
  const float* mb3  = (const float*)d_in[8];
  const float* wlw  = (const float*)d_in[9];
  const float* wlb  = (const float*)d_in[10];
  const float* w0w  = (const float*)d_in[11];
  const float* w0b  = (const float*)d_in[12];
  const float* w1w  = (const float*)d_in[13];
  const float* w1b  = (const float*)d_in[14];

  float* hbuf = (float*)d_ws;                          // [N,128] f32
  float* agg  = hbuf + (size_t)N_NODES * 128;          // [N,128] f32

  hipMemsetAsync(agg, 0, (size_t)N_NODES * 128 * sizeof(float), stream);

  const int gemm_blocks = (N_NODES + 63) / 64;         // 64 rows per 4-wave block
  mlp_kernel<<<gemm_blocks, 128, 0, stream>>>(x, mw1, mb1, mw2, mb2, mw3, mb3, hbuf);

  const int scat_blocks = (N_EDGES * 32 + 255) / 256;  // 1 edge per wave
  scatter_kernel<<<scat_blocks, 256, 0, stream>>>(hbuf, ei, agg);

  out_kernel<<<gemm_blocks, 128, 0, stream>>>(agg, hbuf, x,
                                              wlw, wlb, w0w, w0b, w1w, w1b,
                                              (float*)d_out);
}
